// MRT12_Universal_26594437497335
// MI455X (gfx1250) — matrix operand order repaired and checked by reading, not compiled, not run on hardware
//
#include <hip/hip_runtime.h>
#include <hip/hip_bf16.h>
#include <math.h>

// ---------------- model constants ----------------
#define B_  2
#define S_  1024
#define D_  1024
#define D2_ 2048
#define L_  8
#define V_  32000
#define M_  (B_ * S_)   // 2048 rows of activations

typedef __attribute__((ext_vector_type(16))) __bf16 v16bf;
typedef __attribute__((ext_vector_type(8)))  __bf16 v8bf;
typedef __attribute__((ext_vector_type(8)))  float  v8f;

union V16U { v16bf v; v8bf h[2]; };

// ---------------- helpers ----------------
__device__ __forceinline__ float blockReduceSum(float v, float* red) {
    __syncthreads();                    // make back-to-back calls safe
    red[threadIdx.x] = v;
    __syncthreads();
    for (int off = 128; off > 0; off >>= 1) {
        if ((int)threadIdx.x < off) red[threadIdx.x] += red[threadIdx.x + off];
        __syncthreads();
    }
    return red[0];
}

// ---------------- WMMA GEMM with async global->LDS double buffering --------
// C[M,N] = A[M,K] (bf16 row-major) * Bt[N,K] (bf16, N-major so K contiguous)
// block = 256 threads = 8 waves; block tile 128(M) x 64(N); K step 32.
// Per K-step the block stages A(128x32) and B(64x32) tiles in LDS via
// GLOBAL_LOAD_ASYNC_TO_LDS_B128 (ASYNCcnt), double buffered.
// LDS rows padded to 80B -> conflict-free ds_load_b128 fragment reads.
// M % 128 == 0, N % 64 == 0, K % 32 == 0 for all call sites.
#define EPI_STORE_F32     0
#define EPI_BIAS_SILU_BF  1
#define EPI_BIAS_F32      2
#define EPI_ADD_F32       3

#define A_BUF_BYTES 10240u           // 128 rows * 80B
#define B_BUF_BYTES 5120u            // 64 rows * 80B
#define LDS_BYTES   (2u * A_BUF_BYTES + 2u * B_BUF_BYTES)   // 30720

#define ASYNC_CP16(lds, gp) \
    asm volatile("global_load_async_to_lds_b128 %0, %1, off" \
                 :: "v"(lds), "v"(gp) : "memory")

template<int EPI>
__global__ __launch_bounds__(256)
void gemm_bf16_wmma(const __bf16* __restrict__ A, const __bf16* __restrict__ Bt,
                    const float* __restrict__ bias,
                    float* __restrict__ Cf, __bf16* __restrict__ Cb,
                    float* __restrict__ Acc,
                    int M, int N, int K)
{
    __shared__ char smem[LDS_BYTES] __attribute__((aligned(128)));

    const int tid  = threadIdx.x;
    const int lane = tid & 31;
    const int wave = tid >> 5;                 // 0..7 (stacked along M)
    const int mblk = blockIdx.y * 128;
    const int nblk = blockIdx.x * 64;
    const int ml   = lane & 15;
    const int kh   = lane >> 4;                // 0/1 half-wave

    // ---- async copy plumbing: 16B chunks; 4 chunks per 64B (32-elem) row ----
    const int ra0 = tid >> 2;                  // A rows 0..63   (chunk = tid)
    const int ra1 = (tid >> 2) + 64;           // A rows 64..127 (chunk = tid+256)
    const int rb  = tid >> 2;                  // B rows 0..63
    const int sub = tid & 3;                   // 16B sub-chunk within row

    const __bf16* gA0 = A  + (size_t)(mblk + ra0) * K + sub * 8;
    const __bf16* gA1 = A  + (size_t)(mblk + ra1) * K + sub * 8;
    const __bf16* gB  = Bt + (size_t)(nblk + rb ) * K + sub * 8;

    // generic->LDS offset: low 32 bits of a shared-memory generic address
    const unsigned lbase = (unsigned)(uintptr_t)&smem[0];
    const unsigned lA0a = lbase + (unsigned)(ra0 * 80 + sub * 16);
    const unsigned lA0b = lA0a + A_BUF_BYTES;
    const unsigned lA1a = lbase + (unsigned)(ra1 * 80 + sub * 16);
    const unsigned lA1b = lA1a + A_BUF_BYTES;
    const unsigned lBa  = lbase + 2u * A_BUF_BYTES + (unsigned)(rb * 80 + sub * 16);
    const unsigned lBb  = lBa + B_BUF_BYTES;

    // fragment read offsets within a buffer
    const unsigned aoff = (unsigned)((wave * 16 + ml) * 80 + kh * 32);
    const unsigned b0o  = (unsigned)(ml * 80 + kh * 32);
    const unsigned b1o  = b0o + 16u * 80u;
    const unsigned b2o  = b0o + 32u * 80u;
    const unsigned b3o  = b0o + 48u * 80u;

    // prologue: stage k=0 tile into buffer A
    ASYNC_CP16(lA0a, gA0);
    ASYNC_CP16(lA1a, gA1);
    ASYNC_CP16(lBa,  gB);

    v8f acc0 = {}, acc1 = {}, acc2 = {}, acc3 = {};
    const int nk = K >> 5;

    for (int i = 0; i < nk; ++i) {
        const int cur = i & 1;
        if (i + 1 < nk) {
            gA0 += 32; gA1 += 32; gB += 32;              // advance K by 32 elems (64B)
            if ((i + 1) & 1) { ASYNC_CP16(lA0b, gA0); ASYNC_CP16(lA1b, gA1); ASYNC_CP16(lBb, gB); }
            else             { ASYNC_CP16(lA0a, gA0); ASYNC_CP16(lA1a, gA1); ASYNC_CP16(lBa, gB); }
            asm volatile("s_wait_asynccnt 0x3" ::: "memory");  // current tile done, next in flight
        } else {
            asm volatile("s_wait_asynccnt 0x0" ::: "memory");
        }
        __syncthreads();                                  // tile visible to all waves

        const char* ab = smem + (cur ? A_BUF_BYTES : 0u);
        const char* bb = smem + 2u * A_BUF_BYTES + (cur ? B_BUF_BYTES : 0u);

        V16U a, b0, b1, b2, b3;
        a.h[0]  = *(const v8bf*)(ab + aoff);
        a.h[1]  = *(const v8bf*)(ab + aoff + 16);
        b0.h[0] = *(const v8bf*)(bb + b0o);  b0.h[1] = *(const v8bf*)(bb + b0o + 16);
        b1.h[0] = *(const v8bf*)(bb + b1o);  b1.h[1] = *(const v8bf*)(bb + b1o + 16);
        b2.h[0] = *(const v8bf*)(bb + b2o);  b2.h[1] = *(const v8bf*)(bb + b2o + 16);
        b3.h[0] = *(const v8bf*)(bb + b3o);  b3.h[1] = *(const v8bf*)(bb + b3o + 16);

        acc0 = __builtin_amdgcn_wmma_f32_16x16x32_bf16(false, a.v, false, b0.v, (short)0, acc0, false, false);
        acc1 = __builtin_amdgcn_wmma_f32_16x16x32_bf16(false, a.v, false, b1.v, (short)0, acc1, false, false);
        acc2 = __builtin_amdgcn_wmma_f32_16x16x32_bf16(false, a.v, false, b2.v, (short)0, acc2, false, false);
        acc3 = __builtin_amdgcn_wmma_f32_16x16x32_bf16(false, a.v, false, b3.v, (short)0, acc3, false, false);

        __syncthreads();                                  // all reads done -> buffer reusable
    }

    v8f accs[4] = {acc0, acc1, acc2, acc3};
    #pragma unroll
    for (int t = 0; t < 4; ++t) {
        const int n = nblk + t * 16 + ml;
        float bv = 0.0f;
        if (EPI == EPI_BIAS_SILU_BF || EPI == EPI_BIAS_F32) bv = bias[n];
        #pragma unroll
        for (int r = 0; r < 8; ++r) {
            const int m = mblk + wave * 16 + kh * 8 + r;  // C/D: VGPR r -> M = r + 8*(lane>>4)
            const size_t idx = (size_t)m * N + n;
            float c = accs[t][r];
            if (EPI == EPI_STORE_F32) {
                Cf[idx] = c;
            } else if (EPI == EPI_BIAS_SILU_BF) {
                c += bv;
                c = c / (1.0f + __expf(-c));              // silu
                Cb[idx] = (__bf16)c;
            } else if (EPI == EPI_BIAS_F32) {
                Cf[idx] = c + bv;
            } else {                                      // EPI_ADD_F32 (plastic into x)
                Acc[idx] += c;
            }
        }
    }
}

// ---------------- weight conversion ----------------
// fp32 [mat][K][N] -> bf16 [mat][N][K]  (tiled 32x32 transpose)
__global__ __launch_bounds__(256)
void convert_transpose(const float* __restrict__ in, __bf16* __restrict__ out,
                       int K, int N)
{
    __shared__ float tile[32][33];
    const int mat = blockIdx.z;
    in  += (size_t)mat * K * N;
    out += (size_t)mat * N * K;
    const int kb = blockIdx.y * 32, nb = blockIdx.x * 32;
    const int tx = threadIdx.x & 31, ty = threadIdx.x >> 5;   // 32x8
    #pragma unroll
    for (int i = 0; i < 32; i += 8)
        tile[ty + i][tx] = in[(size_t)(kb + ty + i) * N + nb + tx];
    __syncthreads();
    #pragma unroll
    for (int i = 0; i < 32; i += 8)
        out[(size_t)(nb + ty + i) * K + kb + tx] = (__bf16)tile[tx][ty + i];
}

__global__ void convert_bf16(const float* __restrict__ in, __bf16* __restrict__ out, size_t n)
{
    size_t i = (size_t)blockIdx.x * blockDim.x + threadIdx.x;
    size_t stride = (size_t)gridDim.x * blockDim.x;
    for (; i < n; i += stride) out[i] = (__bf16)in[i];
}

// ---------------- embedding + L2 normalize ----------------
__global__ __launch_bounds__(256)
void embed_norm(const int* __restrict__ ids, const float* __restrict__ emb,
                const float* __restrict__ pos, float* __restrict__ x,
                __bf16* __restrict__ xb)
{
    __shared__ float red[256];
    __shared__ float scs;
    const int row = blockIdx.x;            // b*S + s
    const int s   = row % S_;
    const int id  = ids[row];
    float v[4], ss = 0.0f;
    #pragma unroll
    for (int i = 0; i < 4; ++i) {
        const int d = threadIdx.x + i * 256;
        v[i] = emb[(size_t)id * D_ + d] + pos[(size_t)s * D_ + d];
        ss += v[i] * v[i];
    }
    ss = blockReduceSum(ss, red);
    if (threadIdx.x == 0) scs = 1.0f / fmaxf(sqrtf(ss), 1e-12f);
    __syncthreads();
    const float scale = scs;
    #pragma unroll
    for (int i = 0; i < 4; ++i) {
        const int d = threadIdx.x + i * 256;
        const float y = v[i] * scale;
        x [(size_t)row * D_ + d] = y;
        xb[(size_t)row * D_ + d] = (__bf16)y;
    }
}

// ---------------- fused lerp-scan + causal depthwise conv(k=3) ----------------
// one thread per (b,d) channel; recurrence in registers, conv taps in registers
__global__ void scan_conv(const float* __restrict__ res,
                          const float* __restrict__ cw, const float* __restrict__ cb,
                          __bf16* __restrict__ zb)
{
    const int tid = blockIdx.x * blockDim.x + threadIdx.x;
    if (tid >= B_ * D_) return;
    const int b = tid / D_, d = tid % D_;
    const float w0 = cw[d * 3 + 0], w1 = cw[d * 3 + 1], w2 = cw[d * 3 + 2];
    const float bias = cb[d];
    float h2 = 0.0f, h1 = 0.0f;
    const float* base = res + (size_t)b * S_ * D2_ + d;
    __bf16* zbase = zb + (size_t)b * S_ * D_ + d;
    #pragma unroll 4
    for (int t = 0; t < S_; ++t) {
        const float theta = base[(size_t)t * D2_];
        const float alpha = base[(size_t)t * D2_ + D_];
        const float vv = tanhf(theta) * 3.14159265358979323846f;
        const float a  = 1.0f / (1.0f + __expf(-alpha));
        const float h  = h1 + a * (vv - h1);          // (1-a)*h_prev + a*v
        const float hb = w0 * h2 + w1 * h1 + w2 * h + bias;
        zbase[(size_t)t * D_] = (__bf16)(h + hb);
        h2 = h1; h1 = h;
    }
}

// ---------------- rmsnorm(delta) + residual add ----------------
__global__ __launch_bounds__(256)
void rmsnorm_residual(const float* __restrict__ hf, const float* __restrict__ nw,
                      float* __restrict__ x, __bf16* __restrict__ db,
                      __bf16* __restrict__ xb, int writeXB)
{
    __shared__ float red[256];
    const int row = blockIdx.x;
    const float* hr = hf + (size_t)row * D_;
    float v[4], ss = 0.0f;
    #pragma unroll
    for (int i = 0; i < 4; ++i) {
        const int d = threadIdx.x + i * 256;
        v[i] = hr[d];
        ss += v[i] * v[i];
    }
    ss = blockReduceSum(ss, red);
    const float scale = rsqrtf(ss * (1.0f / D_) + 1e-8f);
    #pragma unroll
    for (int i = 0; i < 4; ++i) {
        const int d = threadIdx.x + i * 256;
        const size_t idx = (size_t)row * D_ + d;
        const float delta = v[i] * scale * nw[d];
        const float xn = x[idx] + delta;
        x[idx] = xn;
        db[idx] = (__bf16)delta;
        if (writeXB) xb[idx] = (__bf16)xn;
    }
}

// ---------------- final: rmsnorm(w) then rmsnorm*0.8, emit bf16 ----------------
__global__ __launch_bounds__(256)
void final_norm(const float* __restrict__ x, const float* __restrict__ fw,
                __bf16* __restrict__ xb)
{
    __shared__ float red[256];
    const int row = blockIdx.x;
    const float* xr = x + (size_t)row * D_;
    float v[4], ss = 0.0f;
    #pragma unroll
    for (int i = 0; i < 4; ++i) {
        const int d = threadIdx.x + i * 256;
        v[i] = xr[d];
        ss += v[i] * v[i];
    }
    ss = blockReduceSum(ss, red);
    const float r1 = rsqrtf(ss * (1.0f / D_) + 1e-8f);
    float y[4], ss2 = 0.0f;
    #pragma unroll
    for (int i = 0; i < 4; ++i) {
        const int d = threadIdx.x + i * 256;
        y[i] = v[i] * r1 * fw[d];
        ss2 += y[i] * y[i];
    }
    ss2 = blockReduceSum(ss2, red);
    const float r2 = rsqrtf(ss2 * (1.0f / D_) + 1e-8f) * 0.8f;
    #pragma unroll
    for (int i = 0; i < 4; ++i) {
        const int d = threadIdx.x + i * 256;
        xb[(size_t)row * D_ + d] = (__bf16)(y[i] * r2);
    }
}

// ---------------- per-row unbiased std normalize + clip (in place on d_out) ----
__global__ __launch_bounds__(256)
void std_clip(float* __restrict__ logits)
{
    __shared__ float red[256];
    const int row = blockIdx.x;
    float* p = logits + (size_t)row * V_;
    float s = 0.0f, s2 = 0.0f;
    for (int i = threadIdx.x; i < V_; i += 256) {
        const float v = p[i];
        s += v; s2 += v * v;
    }
    s  = blockReduceSum(s,  red);
    s2 = blockReduceSum(s2, red);
    const float mean = s * (1.0f / V_);
    float var = (s2 - (float)V_ * mean * mean) / (float)(V_ - 1);
    var = fmaxf(var, 0.0f);
    const float scale = 1.0f / fmaxf(sqrtf(var), 1.0f);
    for (int i = threadIdx.x; i < V_; i += 256) {
        float v = p[i] * scale;
        v = fminf(fmaxf(v, -10.0f), 10.0f);
        p[i] = v;
    }
}

// ---------------- host orchestration ----------------
extern "C" void kernel_launch(void* const* d_in, const int* in_sizes, int n_in,
                              void* d_out, int out_size, void* d_ws, size_t ws_size,
                              hipStream_t stream)
{
    (void)in_sizes; (void)n_in; (void)out_size; (void)ws_size;
    const int*   ids = (const int*)  d_in[0];
    const float* emb = (const float*)d_in[1];
    const float* pos = (const float*)d_in[2];
    const float* kw  = (const float*)d_in[3];
    const float* cw  = (const float*)d_in[4];
    const float* cbv = (const float*)d_in[5];
    const float* w1  = (const float*)d_in[6];
    const float* b1  = (const float*)d_in[7];
    const float* w2  = (const float*)d_in[8];
    const float* b2  = (const float*)d_in[9];
    const float* nw  = (const float*)d_in[10];
    const float* pl  = (const float*)d_in[11];
    const float* fw  = (const float*)d_in[12];
    float* out = (float*)d_out;

    char* ws = (char*)d_ws;
    size_t off = 0;
    auto alloc = [&](size_t bytes) -> void* {
        void* p = ws + off;
        off = (off + bytes + 255) & ~(size_t)255;
        return p;
    };

    float*  x      = (float*) alloc((size_t)M_ * D_  * 4);
    __bf16* xb     = (__bf16*)alloc((size_t)M_ * D_  * 2);
    float*  resbuf = (float*) alloc((size_t)M_ * D2_ * 4);
    __bf16* zb     = (__bf16*)alloc((size_t)M_ * D_  * 2);
    __bf16* a1b    = (__bf16*)alloc((size_t)M_ * D2_ * 2);
    float*  hfbuf  = (float*) alloc((size_t)M_ * D_  * 4);
    __bf16* db     = (__bf16*)alloc((size_t)M_ * D_  * 2);
    __bf16* xfb    = (__bf16*)alloc((size_t)M_ * D_  * 2);
    __bf16* kwT    = (__bf16*)alloc((size_t)L_ * D2_ * D_  * 2);
    __bf16* w1T    = (__bf16*)alloc((size_t)L_ * D2_ * D_  * 2);
    __bf16* w2T    = (__bf16*)alloc((size_t)L_ * D_  * D2_ * 2);
    __bf16* plT    = (__bf16*)alloc((size_t)4  * D_  * D_  * 2);
    __bf16* embb   = (__bf16*)alloc((size_t)V_ * D_  * 2);

    // ---- one-time weight conversion (deterministic, every call) ----
    convert_transpose<<<dim3(D2_/32, D_/32,  L_), 256, 0, stream>>>(kw, kwT, D_,  D2_);
    convert_transpose<<<dim3(D2_/32, D_/32,  L_), 256, 0, stream>>>(w1, w1T, D_,  D2_);
    convert_transpose<<<dim3(D_/32,  D2_/32, L_), 256, 0, stream>>>(w2, w2T, D2_, D_ );
    convert_transpose<<<dim3(D_/32,  D_/32,  4 ), 256, 0, stream>>>(pl, plT, D_,  D_ );
    convert_bf16<<<2048, 256, 0, stream>>>(emb, embb, (size_t)V_ * D_);

    // ---- embed + L2 normalize ----
    embed_norm<<<M_, 256, 0, stream>>>(ids, emb, pos, x, xb);

    // ---- layers ----
    for (int l = 0; l < L_; ++l) {
        // res = x @ kw  [2048 x 2048], K=1024
        gemm_bf16_wmma<EPI_STORE_F32><<<dim3(D2_/64, M_/128), 256, 0, stream>>>(
            xb, kwT + (size_t)l * D2_ * D_, nullptr, resbuf, nullptr, nullptr,
            M_, D2_, D_);
        // h = lerp_scan(tanh*pi, sigmoid); z = h + causal depthwise conv
        scan_conv<<<(B_ * D_) / 256, 256, 0, stream>>>(
            resbuf, cw + (size_t)l * D_ * 3, cbv + (size_t)l * D_, zb);
        // a1 = silu(z @ w1 + b1) -> bf16 [2048 x 2048], K=1024
        gemm_bf16_wmma<EPI_BIAS_SILU_BF><<<dim3(D2_/64, M_/128), 256, 0, stream>>>(
            zb, w1T + (size_t)l * D2_ * D_, b1 + (size_t)l * D2_, nullptr, a1b, nullptr,
            M_, D2_, D_);
        // hf = a1 @ w2 + b2 -> f32 [2048 x 1024], K=2048
        gemm_bf16_wmma<EPI_BIAS_F32><<<dim3(D_/64, M_/128), 256, 0, stream>>>(
            a1b, w2T + (size_t)l * D_ * D2_, b2 + (size_t)l * D_, hfbuf, nullptr, nullptr,
            M_, D_, D2_);
        // delta = rmsnorm(hf, nw); x += delta
        const int plastic = (l >= L_ - 4);
        rmsnorm_residual<<<M_, 256, 0, stream>>>(
            hfbuf, nw + (size_t)l * D_, x, db, xb, plastic ? 0 : 1);
        if (plastic) {
            // x += delta @ plastic[l-4]   [2048 x 1024], K=1024
            gemm_bf16_wmma<EPI_ADD_F32><<<dim3(D_/64, M_/128), 256, 0, stream>>>(
                db, plT + (size_t)(l - 4) * D_ * D_, nullptr, nullptr, nullptr, x,
                M_, D_, D_);
            convert_bf16<<<1024, 256, 0, stream>>>(x, xb, (size_t)M_ * D_);
        }
    }

    // ---- final double rmsnorm * 0.8 -> bf16 ----
    final_norm<<<M_, 256, 0, stream>>>(x, fw, xfb);

    // ---- logits = x @ embedding^T  [2048 x 32000], K=1024 ----
    gemm_bf16_wmma<EPI_STORE_F32><<<dim3(V_/64, M_/128), 256, 0, stream>>>(
        xfb, embb, nullptr, out, nullptr, nullptr, M_, V_, D_);

    // ---- per-row unbiased std normalize + clip ----
    std_clip<<<M_, 256, 0, stream>>>(out);
}
